// feature_net2_57002805952868
// MI455X (gfx1250) — matrix operand order repaired and checked by reading, not compiled, not run on hardware
//
#include <hip/hip_runtime.h>

typedef __attribute__((ext_vector_type(16))) _Float16 v16h;
typedef __attribute__((ext_vector_type(8)))  float    v8f;
typedef __attribute__((ext_vector_type(4)))  float    f4;
typedef __attribute__((ext_vector_type(8)))  int      v8i;

#define NB   4
#define NN   10000
#define NH   32
#define NF   64        // per-head features f
#define KTOT 128       // in_feat = 2f
#define NO   64        // out features
#define BN   (NB*NN)   // 40000 (b,n) pairs

// ---------------------------------------------------------------------------
// Pre-kernel: pack W (64 x 128 f32, row-major [o][k]) into 16 WMMA B fragments
// (f16) laid out exactly as the per-lane VGPR layout of V_WMMA_F32_16X16X32_F16.
// Fragment f = kt*4 + nt (kt = K-step 0..3, nt = N-tile 0..3).
// B (32x16) 16-bit layout: N col = lane%16; lanes 0-15 -> K = h (h<8), 16+(h-8);
// lanes 16-31 -> K = 8+h (h<8), 24+(h-8).  16 halves stored contiguously per lane.
// ---------------------------------------------------------------------------
__global__ void pack_w_kernel(const float* __restrict__ W,
                              _Float16* __restrict__ wsW) {
  int t = threadIdx.x;                   // 256 threads, 32 halves each
  for (int j = 0; j < 32; ++j) {
    int i    = t * 32 + j;               // 0..8191
    int f    = i >> 9;                   // fragment 0..15
    int rem  = i & 511;
    int lane = rem >> 4;                 // 0..31
    int h    = rem & 15;                 // half index within v16h
    int kt = f >> 2, nt = f & 3;
    int o  = nt * 16 + (lane & 15);                // output column
    int kl = ((lane >> 4) << 3) + (h & 7) + ((h >> 3) << 4);  // K within 32-step
    int kk = kt * 32 + kl;                         // global K (= W column)
    wsW[f * 512 + lane * 16 + h] = (_Float16)W[o * KTOT + kk];
  }
}

// ---------------------------------------------------------------------------
// Main fused kernel: one wave per (b,n) pair per grid-stride iteration.
//   e = [x - x_v | x]  (32 x 128)  ->  e @ W^T + b  ->  leaky relu -> max_h
// launch_bounds(256,1): allow a large VGPR allocation (B fragments stay
// resident: 128 VGPRs) with zero spilling; WMMA-dense loop tolerates the
// occupancy trade.
// ---------------------------------------------------------------------------
__global__ void __launch_bounds__(256, 1)
feature_net2_kernel(const float* __restrict__ x,
                    const float* __restrict__ bias,
                    const _Float16* __restrict__ wsW,
                    float* __restrict__ out,
                    int totalWaves) {
  const int lane = threadIdx.x & 31;
  const int wid  = blockIdx.x * (blockDim.x >> 5) + (threadIdx.x >> 5);

  // Resident B fragments: 16 x v16h (W already in WMMA layout, f16)
  v16h Bf[16];
#pragma unroll
  for (int f = 0; f < 16; ++f)
    Bf[f] = *(const v16h*)(wsW + f * 512 + lane * 16);

  // Bias per lane column (N = nt*16 + lane%16)
  float bv[4];
#pragma unroll
  for (int nt = 0; nt < 4; ++nt) bv[nt] = bias[nt * 16 + (lane & 15)];

  const int base  = (lane < 16) ? 0 : 8;  // A-layout K base per lane half
  const int row_l = lane & 15;            // row within M-tile
  const int src   = lane & 16;            // broadcast source lane for x_v

  for (int bn = wid; bn < BN; bn += totalWaves) {
    const float* xp = x + (size_t)bn * (NH * NF);

    // Load x and convert straight to f16 A-fragment staging.
    // xh[m][c]: m = M-tile, c = K-half (c0 = 0 / 32).
    // Per-lane element i: K = base + i + (i>=8 ? 8 : 0)  (two 8-float runs).
    v16h xh[2][2];
#pragma unroll
    for (int m = 0; m < 2; ++m) {
      const float* rp = xp + (m * 16 + row_l) * NF;
#pragma unroll
      for (int c = 0; c < 2; ++c) {
        const float* p = rp + c * 32 + base;
        f4 a0 = __builtin_nontemporal_load((const f4*)(p + 0));
        f4 a1 = __builtin_nontemporal_load((const f4*)(p + 4));
        f4 a2 = __builtin_nontemporal_load((const f4*)(p + 16));
        f4 a3 = __builtin_nontemporal_load((const f4*)(p + 20));
        v16h t;
#pragma unroll
        for (int i = 0; i < 4; ++i) {
          t[i]      = (_Float16)a0[i];
          t[4 + i]  = (_Float16)a1[i];
          t[8 + i]  = (_Float16)a2[i];
          t[12 + i] = (_Float16)a3[i];
        }
        xh[m][c] = t;
      }
    }

    // Broadcast row-0 (x_v) K-sets: lane 0 holds the set for lanes 0-15,
    // lane 16 for lanes 16-31.  Shuffle the 8 packed-f16 dwords.
    v16h xvh[2];
#pragma unroll
    for (int c = 0; c < 2; ++c) {
      v8i s = *(v8i*)&xh[0][c];
      v8i d;
#pragma unroll
      for (int i = 0; i < 8; ++i) d[i] = __shfl(s[i], src, 32);
      xvh[c] = *(v16h*)&d;
    }

    // GEMM: K-steps 0,1 use (x - x_v) halves c=0,32 ; K-steps 2,3 use raw x.
    v8f acc[2][4];
#pragma unroll
    for (int m = 0; m < 2; ++m)
#pragma unroll
      for (int nt = 0; nt < 4; ++nt) acc[m][nt] = (v8f){};

#pragma unroll
    for (int kt = 0; kt < 4; ++kt) {
      v16h A0, A1;
      if (kt < 2) {
        // packed f16 subtract (v_pk_add_f16 w/ neg), built just-in-time
#pragma unroll
        for (int i = 0; i < 16; ++i) {
          A0[i] = xh[0][kt][i] - xvh[kt][i];
          A1[i] = xh[1][kt][i] - xvh[kt][i];
        }
      } else {
        A0 = xh[0][kt - 2];
        A1 = xh[1][kt - 2];
      }
#pragma unroll
      for (int nt = 0; nt < 4; ++nt) {
        acc[0][nt] = __builtin_amdgcn_wmma_f32_16x16x32_f16(
            false, A0, false, Bf[kt * 4 + nt], (short)0, acc[0][nt], false, false);
        acc[1][nt] = __builtin_amdgcn_wmma_f32_16x16x32_f16(
            false, A1, false, Bf[kt * 4 + nt], (short)0, acc[1][nt], false, false);
      }
    }

    // Epilogue: +bias, leaky relu (max(y, 0.2y)), max over 32 heads, store.
    // C layout: VGPR r, lanes 0-15: M = r (+8 for lanes 16-31), N = lane%16.
#pragma unroll
    for (int nt = 0; nt < 4; ++nt) {
      float mx = -3.4e38f;
#pragma unroll
      for (int m = 0; m < 2; ++m)
#pragma unroll
        for (int r = 0; r < 8; ++r) {
          float y = acc[m][nt][r] + bv[nt];
          y = fmaxf(y, 0.2f * y);        // leaky relu, slope 0.2
          mx = fmaxf(mx, y);
        }
      // combine row-halves: lane L holds rows {0-7,16-23} or {8-15,24-31}
      mx = fmaxf(mx, __shfl_xor(mx, 16, 32));
      if (lane < 16)
        __builtin_nontemporal_store(mx, &out[(size_t)bn * NO + nt * 16 + lane]);
    }
  }
}

extern "C" void kernel_launch(void* const* d_in, const int* in_sizes, int n_in,
                              void* d_out, int out_size, void* d_ws, size_t ws_size,
                              hipStream_t stream) {
  const float* x  = (const float*)d_in[0];   // (4,10000,32,64) f32
  const float* W  = (const float*)d_in[1];   // (64,128) f32
  const float* b  = (const float*)d_in[2];   // (64,)   f32
  float* out      = (float*)d_out;           // (4,1,10000,64) f32
  _Float16* wsW   = (_Float16*)d_ws;         // needs 16 KB scratch

  // Pack W into WMMA B-fragment layout (deterministic, re-done every call).
  pack_w_kernel<<<1, 256, 0, stream>>>(W, wsW);

  const int blocks = 1250, threads = 256;            // 10000 waves
  const int totalWaves = blocks * (threads / 32);    // 4 pairs per wave
  feature_net2_kernel<<<blocks, threads, 0, stream>>>(x, b, wsW, out, totalWaves);
}
